// GraphTransformer_6657199309232
// MI455X (gfx1250) — matrix operand order repaired and checked by reading, not compiled, not run on hardware
//
#include <hip/hip_runtime.h>

typedef _Float16 h8   __attribute__((ext_vector_type(8)));
typedef _Float16 h4   __attribute__((ext_vector_type(4)));
typedef _Float16 v16h __attribute__((ext_vector_type(16)));
typedef float    v8f  __attribute__((ext_vector_type(8)));

#define N_NODES    50000
#define N_EDGES    800000
#define DIM        128
#define N_LAYERS   3
#define NUM_GRAPHS 512

// ---------------------------------------------------------------------------
// helpers: order-preserving float <-> uint mapping (for atomic scatter-max)
// ---------------------------------------------------------------------------
__device__ __forceinline__ unsigned fflip(float f) {
    unsigned u = __float_as_uint(f);
    return (u >> 31) ? ~u : (u | 0x80000000u);
}
__device__ __forceinline__ float funflip(unsigned u) {
    unsigned v = (u >> 31) ? (u & 0x7FFFFFFFu) : ~u;
    return __uint_as_float(v);
}

// ---------------------------------------------------------------------------
// f32 -> f16 conversion, float4-vectorized (n must be a multiple of 4)
// ---------------------------------------------------------------------------
__global__ __launch_bounds__(256)
void f32_to_f16_v4(const float* __restrict__ in, _Float16* __restrict__ out, int n4) {
    int i = blockIdx.x * 256 + threadIdx.x;
    if (i < n4) {
        float4 f = ((const float4*)in)[i];
        h4 h;
        h[0] = (_Float16)f.x; h[1] = (_Float16)f.y;
        h[2] = (_Float16)f.z; h[3] = (_Float16)f.w;
        ((h4*)out)[i] = h;
    }
}

// ---------------------------------------------------------------------------
// Fused 4-way GEMM: out_w = xh @ W_w^T + b_w for w in {q,k,v,skip}
// One wave per (16-node tile, matrix); 16x128 output per wave via 8 WMMA
// column tiles, K-loop of 4 (K=32 per v_wmma_f32_16x16x32_f16).
// Per K-step, all 8 B fragments are loaded into distinct registers first
// (one load clause), then 8 WMMAs issue with only partial loadcnt waits.
// ---------------------------------------------------------------------------
__global__ __launch_bounds__(128)
void gemm4(const _Float16* __restrict__ xh,
           const _Float16* __restrict__ w0, const _Float16* __restrict__ w1,
           const _Float16* __restrict__ w2, const _Float16* __restrict__ w3,
           const float* __restrict__ b0, const float* __restrict__ b1,
           const float* __restrict__ b2, const float* __restrict__ b3,
           float* __restrict__ o0, float* __restrict__ o1,
           float* __restrict__ o2, float* __restrict__ o3) {
    const int wave = threadIdx.x >> 5;
    const int lane = threadIdx.x & 31;
    const _Float16* W    = (wave == 0) ? w0 : (wave == 1) ? w1 : (wave == 2) ? w2 : w3;
    const float*    bias = (wave == 0) ? b0 : (wave == 1) ? b1 : (wave == 2) ? b2 : b3;
    float*          out  = (wave == 0) ? o0 : (wave == 1) ? o1 : (wave == 2) ? o2 : o3;

    const int row0 = blockIdx.x * 16;
    const int m    = lane & 15;
    const int hi   = lane >> 4;

    v8f acc[8];
#pragma unroll
    for (int nt = 0; nt < 8; ++nt)
#pragma unroll
        for (int r = 0; r < 8; ++r) acc[nt][r] = 0.0f;

    const _Float16* abase = xh + (size_t)(row0 + m) * DIM + hi * 8;
    const _Float16* bbase = W + (size_t)m * DIM + hi * 8;

#pragma unroll
    for (int kb = 0; kb < 4; ++kb) {
        // A fragment: lane<16 holds K {0..7,16..23} of row M=lane,
        //             lane>=16 holds K {8..15,24..31} of row M=lane-16.
        h8 alo = *(const h8*)(abase + kb * 32);
        h8 ahi = *(const h8*)(abase + kb * 32 + 16);
        v16h A;
#pragma unroll
        for (int i = 0; i < 8; ++i) { A[i] = alo[i]; A[i + 8] = ahi[i]; }

        // All 8 B fragments for this K-step, distinct registers.
        // B layout: lane holds column N=lane&15; lanes 0-15 K={0..7,16..23},
        // lanes 16-31 K={8..15,24..31} (pairs per VGPR).
        v16h B[8];
#pragma unroll
        for (int nt = 0; nt < 8; ++nt) {
            const _Float16* brow = bbase + (size_t)(nt * 16) * DIM + kb * 32;
            h8 blo = *(const h8*)(brow);
            h8 bhi = *(const h8*)(brow + 16);
#pragma unroll
            for (int i = 0; i < 8; ++i) { B[nt][i] = blo[i]; B[nt][i + 8] = bhi[i]; }
        }

#pragma unroll
        for (int nt = 0; nt < 8; ++nt)
            acc[nt] = __builtin_amdgcn_wmma_f32_16x16x32_f16(
                false, A, false, B[nt], (short)0, acc[nt], false, false);
    }

    const int mo = hi * 8;
#pragma unroll
    for (int nt = 0; nt < 8; ++nt) {
        float bv = bias[nt * 16 + m];
#pragma unroll
        for (int r = 0; r < 8; ++r)
            out[(size_t)(row0 + mo + r) * DIM + nt * 16 + m] = acc[nt][r] + bv;
    }
}

// ---------------------------------------------------------------------------
// per-layer node-state init: amax = flip(-inf), denom = 0
// ---------------------------------------------------------------------------
__global__ __launch_bounds__(256)
void init_nodes(unsigned* __restrict__ amax, float* __restrict__ denom) {
    int i = blockIdx.x * 256 + threadIdx.x;
    if (i < N_NODES) { amax[i] = 0x007FFFFFu; denom[i] = 0.0f; }
}

// ---------------------------------------------------------------------------
// alpha[e] = <q[dst], k[src]> * scale ; scatter-max into amax[dst]
// one wave per edge: lane loads float4 (coalesced 128-float rows)
// ---------------------------------------------------------------------------
__global__ __launch_bounds__(256)
void edge_alpha(const float* __restrict__ q, const float* __restrict__ k,
                const long long* __restrict__ src, const long long* __restrict__ dst,
                float* __restrict__ alpha, unsigned* __restrict__ amax) {
    int e = blockIdx.x * 8 + (threadIdx.x >> 5);
    if (e >= N_EDGES) return;
    int lane = threadIdx.x & 31;
    long long s = src[e], d = dst[e];
    float4 a = ((const float4*)(q + (size_t)d * DIM))[lane];
    float4 b = ((const float4*)(k + (size_t)s * DIM))[lane];
    float t = a.x * b.x + a.y * b.y + a.z * b.z + a.w * b.w;
#pragma unroll
    for (int off = 16; off > 0; off >>= 1) t += __shfl_xor(t, off, 32);
    if (lane == 0) {
        t *= 0.088388347648318447f;   // 1/sqrt(128)
        alpha[e] = t;
        atomicMax(amax + d, fflip(t));
    }
}

// ---------------------------------------------------------------------------
// ex[e] = exp(alpha - amax[dst]) (in place); denom[dst] += ex
// ---------------------------------------------------------------------------
__global__ __launch_bounds__(256)
void edge_exp(const long long* __restrict__ dst, const unsigned* __restrict__ amax,
              float* __restrict__ alpha, float* __restrict__ denom) {
    int e = blockIdx.x * 256 + threadIdx.x;
    if (e >= N_EDGES) return;
    long long d = dst[e];
    float ex = __expf(alpha[e] - funflip(amax[d]));
    alpha[e] = ex;
    unsafeAtomicAdd(denom + d, ex);
}

// ---------------------------------------------------------------------------
// y[dst] += (ex/denom[dst]) * v[src]  -- wave per edge, f32 atomic adds
// ---------------------------------------------------------------------------
__global__ __launch_bounds__(256)
void edge_scatter(const float* __restrict__ alpha, const float* __restrict__ denom,
                  const float* __restrict__ v, const long long* __restrict__ src,
                  const long long* __restrict__ dst, float* __restrict__ y) {
    int e = blockIdx.x * 8 + (threadIdx.x >> 5);
    if (e >= N_EDGES) return;
    int lane = threadIdx.x & 31;
    long long s = src[e], d = dst[e];
    float w = alpha[e] / denom[d];
    float4 vv = ((const float4*)(v + (size_t)s * DIM))[lane];
    float* yp = y + (size_t)d * DIM + lane * 4;
    unsafeAtomicAdd(yp + 0, w * vv.x);
    unsafeAtomicAdd(yp + 1, w * vv.y);
    unsafeAtomicAdd(yp + 2, w * vv.z);
    unsafeAtomicAdd(yp + 3, w * vv.w);
}

// ---------------------------------------------------------------------------
// x = relu(y) in f32 (kept in y) and f16 (xh, next layer's GEMM input)
// float4-vectorized stream kernel (HBM-bound; 23.3 TB/s path)
// ---------------------------------------------------------------------------
__global__ __launch_bounds__(256)
void relu_cvt_v4(float* __restrict__ y, _Float16* __restrict__ xh, int n4) {
    int i = blockIdx.x * 256 + threadIdx.x;
    if (i < n4) {
        float4 f = ((float4*)y)[i];
        f.x = fmaxf(f.x, 0.0f); f.y = fmaxf(f.y, 0.0f);
        f.z = fmaxf(f.z, 0.0f); f.w = fmaxf(f.w, 0.0f);
        ((float4*)y)[i] = f;
        h4 h;
        h[0] = (_Float16)f.x; h[1] = (_Float16)f.y;
        h[2] = (_Float16)f.z; h[3] = (_Float16)f.w;
        ((h4*)xh)[i] = h;
    }
}

// ---------------------------------------------------------------------------
// pooling
// ---------------------------------------------------------------------------
__global__ __launch_bounds__(256)
void zero_pool(float* __restrict__ out, float* __restrict__ counts) {
    int i = blockIdx.x * 256 + threadIdx.x;
    if (i < NUM_GRAPHS * DIM) out[i] = 0.0f;
    if (i < NUM_GRAPHS) counts[i] = 0.0f;
}

__global__ __launch_bounds__(256)
void pool_scatter(const float* __restrict__ y, const long long* __restrict__ batch,
                  float* __restrict__ out, float* __restrict__ counts) {
    int nidx = blockIdx.x * 8 + (threadIdx.x >> 5);
    if (nidx >= N_NODES) return;
    int lane = threadIdx.x & 31;
    long long g = batch[nidx];
    float4 xv = ((const float4*)(y + (size_t)nidx * DIM))[lane];
    float* op = out + (size_t)g * DIM + lane * 4;
    unsafeAtomicAdd(op + 0, xv.x);
    unsafeAtomicAdd(op + 1, xv.y);
    unsafeAtomicAdd(op + 2, xv.z);
    unsafeAtomicAdd(op + 3, xv.w);
    if (lane == 0) unsafeAtomicAdd(counts + g, 1.0f);
}

__global__ __launch_bounds__(256)
void pool_div(float* __restrict__ out, const float* __restrict__ counts) {
    int i = blockIdx.x * 256 + threadIdx.x;
    if (i < NUM_GRAPHS * DIM) out[i] /= fmaxf(counts[i >> 7], 1.0f);
}

// ---------------------------------------------------------------------------
// host orchestration
// ---------------------------------------------------------------------------
extern "C" void kernel_launch(void* const* d_in, const int* in_sizes, int n_in,
                              void* d_out, int out_size, void* d_ws, size_t ws_size,
                              hipStream_t stream) {
    const float*     x     = (const float*)d_in[0];
    const long long* edge  = (const long long*)d_in[1];
    const long long* src   = edge;
    const long long* dst   = edge + N_EDGES;
    const long long* batch = (const long long*)d_in[2];
    const float* Wq = (const float*)d_in[3];
    const float* bq = (const float*)d_in[4];
    const float* Wk = (const float*)d_in[5];
    const float* bk = (const float*)d_in[6];
    const float* Wv = (const float*)d_in[7];
    const float* bv = (const float*)d_in[8];
    const float* Ws = (const float*)d_in[9];
    const float* bs = (const float*)d_in[10];

    // workspace carve-up
    char*  ws  = (char*)d_ws;
    size_t off = 0;
    auto alloc = [&](size_t bytes) -> void* {
        void* p = ws + off;
        off = (off + bytes + 255) & ~(size_t)255;
        return p;
    };
    _Float16* xh    = (_Float16*)alloc((size_t)N_NODES * DIM * 2);
    float*    q     = (float*)alloc((size_t)N_NODES * DIM * 4);
    float*    k     = (float*)alloc((size_t)N_NODES * DIM * 4);
    float*    v     = (float*)alloc((size_t)N_NODES * DIM * 4);
    float*    y     = (float*)alloc((size_t)N_NODES * DIM * 4);
    float*    alpha = (float*)alloc((size_t)N_EDGES * 4);
    unsigned* amax  = (unsigned*)alloc((size_t)N_NODES * 4);
    float*    denom = (float*)alloc((size_t)N_NODES * 4);
    float*    cnts  = (float*)alloc((size_t)NUM_GRAPHS * 4);
    _Float16* whq   = (_Float16*)alloc((size_t)N_LAYERS * DIM * DIM * 2);
    _Float16* whk   = (_Float16*)alloc((size_t)N_LAYERS * DIM * DIM * 2);
    _Float16* whv   = (_Float16*)alloc((size_t)N_LAYERS * DIM * DIM * 2);
    _Float16* whs   = (_Float16*)alloc((size_t)N_LAYERS * DIM * DIM * 2);

    const int ND4 = N_NODES * DIM / 4;        // 1,600,000 float4s
    const int WN4 = N_LAYERS * DIM * DIM / 4; // 12,288 float4s

    // one-time conversions
    f32_to_f16_v4<<<(ND4 + 255) / 256, 256, 0, stream>>>(x, xh, ND4);
    f32_to_f16_v4<<<(WN4 + 255) / 256, 256, 0, stream>>>(Wq, whq, WN4);
    f32_to_f16_v4<<<(WN4 + 255) / 256, 256, 0, stream>>>(Wk, whk, WN4);
    f32_to_f16_v4<<<(WN4 + 255) / 256, 256, 0, stream>>>(Wv, whv, WN4);
    f32_to_f16_v4<<<(WN4 + 255) / 256, 256, 0, stream>>>(Ws, whs, WN4);

    for (int l = 0; l < N_LAYERS; ++l) {
        const size_t wo = (size_t)l * DIM * DIM;
        const size_t bo = (size_t)l * DIM;
        // q,k,v and y = x@Ws^T + bs (skip) in one fused WMMA launch
        gemm4<<<N_NODES / 16, 128, 0, stream>>>(
            xh, whq + wo, whk + wo, whv + wo, whs + wo,
            bq + bo, bk + bo, bv + bo, bs + bo,
            q, k, v, y);
        init_nodes<<<(N_NODES + 255) / 256, 256, 0, stream>>>(amax, denom);
        edge_alpha<<<N_EDGES / 8, 256, 0, stream>>>(q, k, src, dst, alpha, amax);
        edge_exp<<<N_EDGES / 256, 256, 0, stream>>>(dst, amax, alpha, denom);
        edge_scatter<<<N_EDGES / 8, 256, 0, stream>>>(alpha, denom, v, src, dst, y);
        relu_cvt_v4<<<ND4 / 256, 256, 0, stream>>>(y, xh, ND4);
    }

    // global mean pool into d_out
    float* out = (float*)d_out;
    zero_pool<<<(NUM_GRAPHS * DIM + 255) / 256, 256, 0, stream>>>(out, cnts);
    pool_scatter<<<N_NODES / 8, 256, 0, stream>>>(y, batch, out, cnts);
    pool_div<<<(NUM_GRAPHS * DIM + 255) / 256, 256, 0, stream>>>(out, cnts);
}
